// RNN_69217692942804
// MI455X (gfx1250) — matrix operand order repaired and checked by reading, not compiled, not run on hardware
//
#include <hip/hip_runtime.h>
#include <hip/hip_bf16.h>

typedef __attribute__((ext_vector_type(16))) __bf16 v16bf;
typedef __attribute__((ext_vector_type(8)))  __bf16 v8bf;
typedef __attribute__((ext_vector_type(8)))  float  v8f;

#define DEV_INLINE __device__ __forceinline__

constexpr int Bb = 64, Ss = 512, Ii = 512, Hh = 1024, Oo = 512;
constexpr int NBLK = 64;  // persistent blocks per layer kernel (1 hidden tile each)

// A-matrix fragment (16x32 bf16, M x K): lane half h holds K = h*8+{0..7} and 16+h*8+{0..7}
DEV_INLINE v16bf load_fragA(const __bf16* p, int half) {
  v8bf lo = *(const v8bf*)(p + half * 8);
  v8bf hi = *(const v8bf*)(p + 16 + half * 8);
  v16bf r;
#pragma unroll
  for (int i = 0; i < 8; ++i) { r[i] = lo[i]; r[8 + i] = hi[i]; }
  return r;
}

// B-matrix fragment (32x16 bf16, K x N): lane half h holds K = h*8+{0..7} and 16+h*8+{0..7}
DEV_INLINE v16bf load_fragB(const __bf16* p, int half) {
  v8bf lo = *(const v8bf*)(p + half * 8);
  v8bf hi = *(const v8bf*)(p + 16 + half * 8);
  v16bf r;
#pragma unroll
  for (int i = 0; i < 8; ++i) { r[i] = lo[i]; r[8 + i] = hi[i]; }
  return r;
}

DEV_INLINE float fast_sigmoid(float x) { return 1.0f / (1.0f + __expf(-x)); }
DEV_INLINE float fast_tanh(float x) { return 1.0f - 2.0f / (__expf(2.0f * x) + 1.0f); }

// Persistent LSTM layer kernel: one launch runs all S timesteps.
// Grid: 64 blocks x 128 threads (4 wave32). Block b owns hidden cols [16b, 16b+16)
// for all 4 gates; wave w owns batch rows [16w, 16w+16). Weight slice (4x16 rows x K)
// is staged once into LDS (padded rows) and reused for all 512 steps; cell state
// lives in registers for the whole sequence. Steps are separated by a device-wide
// release/acquire spin barrier (64 co-resident workgroups).
template <int KIN>
__global__ __launch_bounds__(128) void lstm_layer_kernel(
    const __bf16* __restrict__ xin,   // [B, S, KIN]  layer input sequence (bf16)
    __bf16* __restrict__ yout,        // [B, S, H]    h history (written at t, read at t-1)
    const __bf16* __restrict__ Wcat,  // [4H, KIN+H]  packed [W_ih ; W_hh] (bf16)
    const float* __restrict__ bias,   // [4H]         b_ih + b_hh
    float* __restrict__ cstate,       // [B, H]       final cell state out (fp32)
    unsigned* __restrict__ bar) {     // grid barrier counter (pre-zeroed)
  constexpr int K = KIN + Hh;
  constexpr int LROW = K + 8;         // padded LDS row, keeps b128 reads spread over banks
  extern __shared__ __bf16 ldsw[];    // [64 * LROW] bf16

  const int tid  = threadIdx.x;
  const int wave = tid >> 5;
  const int lane = tid & 31;
  const int half = lane >> 4;
  const int l16  = lane & 15;
  const int nBase = blockIdx.x * 16;  // hidden tile
  const int mBase = wave * 16;        // batch tile
  const int mRow  = mBase + l16;
  const int nCol  = nBase + l16;

  // ---- Stage this block's weight slice into LDS (16B chunks) ----
  {
    constexpr int C8 = K / 8;  // b128 chunks per row
    for (int idx = tid; idx < 64 * C8; idx += 128) {
      const int row = idx / C8;          // 0..63 = gate*16 + j
      const int c8  = idx % C8;
      const int g = row >> 4, j = row & 15;
      const uint4 v = *(const uint4*)(Wcat + ((size_t)(g * Hh + nBase + j)) * K + c8 * 8);
      *(uint4*)(ldsw + (size_t)row * LROW + c8 * 8) = v;
    }
  }
  __syncthreads();

  const float bi = bias[nCol];
  const float bf = bias[Hh + nCol];
  const float bg = bias[2 * Hh + nCol];
  const float bo = bias[3 * Hh + nCol];

  float creg[8];  // cell state: this lane's 8 (row, nCol) entries, resident all steps
#pragma unroll
  for (int r = 0; r < 8; ++r) creg[r] = 0.0f;

  for (int t = 0; t < Ss; ++t) {
    v8f acc[4];
#pragma unroll
    for (int g = 0; g < 4; ++g)
#pragma unroll
      for (int r = 0; r < 8; ++r) acc[g][r] = 0.0f;

    // Input contribution
    const __bf16* arow = xin + ((size_t)mRow * Ss + t) * KIN;
    for (int kk = 0; kk < KIN; kk += 32) {
      v16bf a = load_fragA(arow + kk, half);
#pragma unroll
      for (int g = 0; g < 4; ++g) {
        v16bf b = load_fragB(ldsw + (size_t)(g * 16 + l16) * LROW + kk, half);
        acc[g] = __builtin_amdgcn_wmma_f32_16x16x32_bf16(
            false, a, false, b, (short)0, acc[g], false, false);
      }
    }
    // Recurrent contribution (h_{-1} == 0)
    if (t > 0) {
      const __bf16* hrow = yout + ((size_t)mRow * Ss + (t - 1)) * Hh;
      for (int kk = 0; kk < Hh; kk += 32) {
        v16bf a = load_fragA(hrow + kk, half);
#pragma unroll
        for (int g = 0; g < 4; ++g) {
          v16bf b = load_fragB(ldsw + (size_t)(g * 16 + l16) * LROW + KIN + kk, half);
          acc[g] = __builtin_amdgcn_wmma_f32_16x16x32_bf16(
              false, a, false, b, (short)0, acc[g], false, false);
        }
      }
    }

    // Gate math + h writeback; C/D layout: VGPR r -> row mBase + r + 8*half, col nCol
#pragma unroll
    for (int r = 0; r < 8; ++r) {
      const int m = mBase + r + 8 * half;
      const float iv = fast_sigmoid(acc[0][r] + bi);
      const float fv = fast_sigmoid(acc[1][r] + bf);
      const float gv = fast_tanh(acc[2][r] + bg);
      const float ov = fast_sigmoid(acc[3][r] + bo);
      creg[r] = fv * creg[r] + iv * gv;
      const float hv = ov * fast_tanh(creg[r]);
      yout[((size_t)m * Ss + t) * Hh + nCol] = (__bf16)hv;
    }

    // Device-wide barrier: h(t) must be visible to every block before step t+1
    __threadfence();
    __syncthreads();
    if (tid == 0) {
      __hip_atomic_fetch_add(bar, 1u, __ATOMIC_RELEASE, __HIP_MEMORY_SCOPE_AGENT);
      const unsigned target = (unsigned)NBLK * (unsigned)(t + 1);
      while (__hip_atomic_load(bar, __ATOMIC_ACQUIRE, __HIP_MEMORY_SCOPE_AGENT) < target)
        __builtin_amdgcn_s_sleep(1);
    }
    __syncthreads();
  }

  // Final cell state for c_n
#pragma unroll
  for (int r = 0; r < 8; ++r) {
    const int m = mBase + r + 8 * half;
    cstate[(size_t)m * Hh + nCol] = creg[r];
  }
}

// out[M, O] = y1[M, H] @ Wfc^T + b ; one wave per 16x16 output tile.
__global__ __launch_bounds__(256) void fc_kernel(
    const __bf16* __restrict__ y1,   // [B*S, H]
    const __bf16* __restrict__ Wfc,  // [O, H]
    const float* __restrict__ bfc,   // [O]
    float* __restrict__ out) {       // [B*S, O]
  const int wave = threadIdx.x >> 5;
  const int lane = threadIdx.x & 31;
  const int half = lane >> 4;
  const int l16  = lane & 15;
  const int gw   = blockIdx.x * 8 + wave;   // 0..65535
  const int mBase = (gw >> 5) * 16;
  const int nBase = (gw & 31) * 16;
  const int mRow  = mBase + l16;
  const int nCol  = nBase + l16;

  v8f acc;
#pragma unroll
  for (int r = 0; r < 8; ++r) acc[r] = 0.0f;

  const __bf16* arow = y1 + (size_t)mRow * Hh;
  const __bf16* brow = Wfc + (size_t)nCol * Hh;
  for (int kk = 0; kk < Hh; kk += 32) {
    v16bf a = load_fragA(arow + kk, half);
    v16bf b = load_fragB(brow + kk, half);
    acc = __builtin_amdgcn_wmma_f32_16x16x32_bf16(
        false, a, false, b, (short)0, acc, false, false);
  }
  const float bb = bfc[nCol];
#pragma unroll
  for (int r = 0; r < 8; ++r) {
    const int m = mBase + r + 8 * half;
    out[(size_t)m * Oo + nCol] = acc[r] + bb;
  }
}

__global__ void f32_to_bf16_kernel(const float* __restrict__ in,
                                   __bf16* __restrict__ out, int n) {
  int i = blockIdx.x * blockDim.x + threadIdx.x;
  if (i < n) out[i] = (__bf16)in[i];
}

// Pack [W_ih ; W_hh] -> bf16 [4H, KIN+H] row-major
__global__ void pack_wcat_kernel(const float* __restrict__ Wih,
                                 const float* __restrict__ Whh,
                                 __bf16* __restrict__ out, int KIN) {
  const int K = KIN + Hh;
  int i = blockIdx.x * blockDim.x + threadIdx.x;
  if (i >= 4 * Hh * K) return;
  int row = i / K, k = i % K;
  float v = (k < KIN) ? Wih[(size_t)row * KIN + k]
                      : Whh[(size_t)row * Hh + (k - KIN)];
  out[i] = (__bf16)v;
}

__global__ void bias_add_kernel(const float* __restrict__ a,
                                const float* __restrict__ b,
                                float* __restrict__ out, int n) {
  int i = blockIdx.x * blockDim.x + threadIdx.x;
  if (i < n) out[i] = a[i] + b[i];
}

__global__ void init_counters_kernel(unsigned* __restrict__ c) {
  if (threadIdx.x < 2) c[threadIdx.x] = 0u;
}

// Emit h_n [2,B,H] and c_n [2,B,H] after out[B*S,O]
__global__ void finalize_kernel(const __bf16* __restrict__ y0,
                                const __bf16* __restrict__ y1,
                                const float* __restrict__ c0,
                                const float* __restrict__ c1,
                                float* __restrict__ out) {
  int i = blockIdx.x * blockDim.x + threadIdx.x;
  if (i >= 2 * Bb * Hh) return;
  int l = i / (Bb * Hh), r = i % (Bb * Hh);
  int b = r / Hh, h = r % Hh;
  const __bf16* y = l ? y1 : y0;
  const float* c = l ? c1 : c0;
  const size_t base = (size_t)Bb * Ss * Oo;
  out[base + i] = (float)y[((size_t)b * Ss + (Ss - 1)) * Hh + h];
  out[base + 2 * Bb * Hh + i] = c[r];
}

extern "C" void kernel_launch(void* const* d_in, const int* in_sizes, int n_in,
                              void* d_out, int out_size, void* d_ws, size_t ws_size,
                              hipStream_t stream) {
  (void)in_sizes; (void)n_in; (void)out_size; (void)ws_size;
  const float* x    = (const float*)d_in[0];
  const float* Wih0 = (const float*)d_in[1];
  const float* Whh0 = (const float*)d_in[2];
  const float* bih0 = (const float*)d_in[3];
  const float* bhh0 = (const float*)d_in[4];
  const float* Wih1 = (const float*)d_in[5];
  const float* Whh1 = (const float*)d_in[6];
  const float* bih1 = (const float*)d_in[7];
  const float* bhh1 = (const float*)d_in[8];
  const float* Wfc  = (const float*)d_in[9];
  const float* bfc  = (const float*)d_in[10];
  float* out = (float*)d_out;

  char* ws = (char*)d_ws;
  size_t off = 0;
  auto alloc = [&](size_t bytes) -> void* {
    void* p = ws + off;
    off = (off + bytes + 255) & ~(size_t)255;
    return p;
  };
  __bf16* xbf   = (__bf16*)alloc((size_t)Bb * Ss * Ii * 2);
  __bf16* y0    = (__bf16*)alloc((size_t)Bb * Ss * Hh * 2);
  __bf16* y1    = (__bf16*)alloc((size_t)Bb * Ss * Hh * 2);
  __bf16* wcat0 = (__bf16*)alloc((size_t)4 * Hh * (Ii + Hh) * 2);
  __bf16* wcat1 = (__bf16*)alloc((size_t)4 * Hh * (Hh + Hh) * 2);
  __bf16* wfcb  = (__bf16*)alloc((size_t)Oo * Hh * 2);
  float*  bias0 = (float*)alloc((size_t)4 * Hh * 4);
  float*  bias1 = (float*)alloc((size_t)4 * Hh * 4);
  float*  c0    = (float*)alloc((size_t)Bb * Hh * 4);
  float*  c1    = (float*)alloc((size_t)Bb * Hh * 4);
  unsigned* bars = (unsigned*)alloc(2 * sizeof(unsigned));

  // One-time per-call conversions / packing (bf16 path for WMMA)
  { int n = Bb * Ss * Ii;       f32_to_bf16_kernel<<<(n + 255) / 256, 256, 0, stream>>>(x, xbf, n); }
  { int n = 4 * Hh * (Ii + Hh); pack_wcat_kernel<<<(n + 255) / 256, 256, 0, stream>>>(Wih0, Whh0, wcat0, Ii); }
  { int n = 4 * Hh * (Hh + Hh); pack_wcat_kernel<<<(n + 255) / 256, 256, 0, stream>>>(Wih1, Whh1, wcat1, Hh); }
  { int n = Oo * Hh;            f32_to_bf16_kernel<<<(n + 255) / 256, 256, 0, stream>>>(Wfc, wfcb, n); }
  { int n = 4 * Hh;             bias_add_kernel<<<(n + 255) / 256, 256, 0, stream>>>(bih0, bhh0, bias0, n); }
  { int n = 4 * Hh;             bias_add_kernel<<<(n + 255) / 256, 256, 0, stream>>>(bih1, bhh1, bias1, n); }
  init_counters_kernel<<<1, 32, 0, stream>>>(bars);

  // Persistent recurrence: one launch per layer, weights LDS-resident, c in registers.
  constexpr size_t lds0 = (size_t)64 * (Ii + Hh + 8) * 2;  // 197,632 B
  constexpr size_t lds1 = (size_t)64 * (Hh + Hh + 8) * 2;  // 263,168 B (<320KB/WG on CDNA5)
  lstm_layer_kernel<Ii><<<NBLK, 128, lds0, stream>>>(xbf, y0, wcat0, bias0, c0, bars + 0);
  lstm_layer_kernel<Hh><<<NBLK, 128, lds1, stream>>>(y0, y1, wcat1, bias1, c1, bars + 1);

  // Output projection: (32768/16)*(512/16) tiles / 8 waves per block = 8192 blocks
  fc_kernel<<<(Bb * Ss / 16) * (Oo / 16) / 8, 256, 0, stream>>>(y1, wfcb, bfc, out);
  finalize_kernel<<<(2 * Bb * Hh + 255) / 256, 256, 0, stream>>>(y0, y1, c0, c1, out);
}
